// Core_10969346474714
// MI455X (gfx1250) — compile-verified, matrix operand order checked by reading, not run on hardware
//
#include <hip/hip_runtime.h>
#include <hip/hip_bf16.h>

// ---------------------------------------------------------------------------
// Types for CDNA5 WMMA (wave32): v_wmma_f32_16x16x32_bf16
//   A frag (k order used here): lane L holds A[M = L%16][K = (L/16)*16 .. +15]  (16 bf16 = 8 VGPRs)
//   B frag (same k order): lane L holds B[K = (L/16)*16 .. +15][N = L%16]  -> load from Bt[N][K]
//   C/D   : VGPR i, lane L -> row (L/16)*8 + i, col L%16    (8 f32 VGPRs)
// ---------------------------------------------------------------------------
typedef __bf16 bf16_t;
typedef __attribute__((ext_vector_type(16))) __bf16 v16bf;
typedef __attribute__((ext_vector_type(8)))  __bf16 v8bf;
typedef __attribute__((ext_vector_type(8)))  float  v8f;

__device__ __forceinline__ v8f wmma_bf16(v16bf a, v16bf b, v8f c) {
  // (neg_a, A, neg_b, B, c_mod, C, reuse_a, reuse_b)
  return __builtin_amdgcn_wmma_f32_16x16x32_bf16(false, a, false, b, (short)0, c,
                                                 false, false);
}

__device__ __forceinline__ v8f v8f_zero() {
  v8f z;
#pragma unroll
  for (int i = 0; i < 8; ++i) z[i] = 0.0f;
  return z;
}

__device__ __forceinline__ bf16_t to_bf16(float f) {
  unsigned u = __builtin_bit_cast(unsigned, f);
  u += 0x7FFFu + ((u >> 16) & 1u);  // round-to-nearest-even
  return __builtin_bit_cast(bf16_t, (unsigned short)(u >> 16));
}

// CDNA5 async copy: GLOBAL_LOAD_ASYNC_TO_LDS_B128 (ASYNCcnt-tracked, §08).
// LDS operand = wave-relative LDS byte address = low 32 bits of the generic
// address of a __shared__ object (ISA §10.2 LDS aperture mapping).
__device__ __forceinline__ void async_copy16(void* lds_dst, const void* gsrc) {
  const unsigned lds = (unsigned)(unsigned long long)(size_t)lds_dst;
  asm volatile("global_load_async_to_lds_b128 %0, %1, off"
               :
               : "v"(lds), "v"(gsrc)
               : "memory");
}
__device__ __forceinline__ void wait_async0() {
  asm volatile("s_wait_asynccnt 0x0" ::: "memory");
}

// Model constants
#define BS_TOK   4096      // B*S
#define S_LEN    2048
#define H_NUM    16
#define D_EMB    1024
#define QKV_OUT  3072
#define N_VOCAB  32000

// ---------------------------------------------------------------------------
// fp32 -> bf16 flat conversion
// ---------------------------------------------------------------------------
__global__ void k_convert_bf16(const float* __restrict__ in,
                               bf16_t* __restrict__ out, size_t n) {
  size_t i = (size_t)blockIdx.x * blockDim.x + threadIdx.x;
  size_t stride = (size_t)gridDim.x * blockDim.x;
  for (; i < n; i += stride) out[i] = to_bf16(in[i]);
}

// fp32 [K][N] -> bf16 [N][K] (transpose through LDS, both sides coalesced)
__global__ void k_transpose_bf16(const float* __restrict__ in,
                                 bf16_t* __restrict__ out, int K, int N) {
  __shared__ float tile[32][33];
  const int nb = blockIdx.x * 32, kb = blockIdx.y * 32;
  const int tx = threadIdx.x & 31, ty = threadIdx.x >> 5;  // 32x8
#pragma unroll
  for (int r = ty; r < 32; r += 8)
    tile[r][tx] = in[(size_t)(kb + r) * N + nb + tx];
  __syncthreads();
#pragma unroll
  for (int r = ty; r < 32; r += 8)
    out[(size_t)(nb + r) * K + kb + tx] = to_bf16(tile[tx][r]);
}

// qbf[r][c] = bf16(qkv[r][c])  for c < 1024 (q slice of fused qkv buffer)
__global__ void k_extract_q(const float* __restrict__ qkv,
                            bf16_t* __restrict__ qbf) {
  size_t i = (size_t)blockIdx.x * blockDim.x + threadIdx.x;
  size_t stride = (size_t)gridDim.x * blockDim.x;
  const size_t n = (size_t)BS_TOK * D_EMB;
  for (; i < n; i += stride) {
    size_t r = i >> 10;
    int c = (int)(i & 1023);
    qbf[i] = to_bf16(qkv[r * QKV_OUT + c]);
  }
}

// Embedding gather: x fp32 + x bf16
__global__ void k_embed(const int* __restrict__ tokens,
                        const float* __restrict__ emb,
                        float* __restrict__ x, bf16_t* __restrict__ xbf) {
  const int bs = blockIdx.x;
  const size_t src = (size_t)tokens[bs] * D_EMB;
  const size_t dst = (size_t)bs * D_EMB;
  for (int c = threadIdx.x; c < D_EMB; c += blockDim.x) {
    float v = emb[src + c];
    x[dst + c] = v;
    xbf[dst + c] = to_bf16(v);
  }
}

// ---------------------------------------------------------------------------
// bf16 WMMA GEMM:  C[M,N] = A[M,K] * Bt[N,K]^T (+ bias[N]) (+ res[M,N])
// block tile 128x128, 8 waves (2x4), wave tile 64x32 -> 4x2 wmma accumulators
// Double-buffered LDS fed by GLOBAL_LOAD_ASYNC_TO_LDS_B128 (ASYNCcnt),
// so tile kb+1 streams into LDS while WMMAs consume tile kb.
// ---------------------------------------------------------------------------
__global__ __launch_bounds__(256) void k_gemm_bf16(
    const bf16_t* __restrict__ A, const bf16_t* __restrict__ Bt,
    const float* __restrict__ bias, const float* __restrict__ res,
    float* __restrict__ C, int M, int N, int K) {
  __shared__ bf16_t sA[2][128][32];
  __shared__ bf16_t sB[2][128][32];

  const int t = threadIdx.x, lane = t & 31, w = t >> 5;
  const int wm = w >> 2, wn = w & 3;               // 2x4 wave grid
  const int m0 = blockIdx.y * 128, n0 = blockIdx.x * 128;
  const int lr = lane & 15, lh = lane >> 4;        // fragment row / K-half

  v8f acc[4][2];
#pragma unroll
  for (int tm = 0; tm < 4; ++tm)
#pragma unroll
    for (int tn = 0; tn < 2; ++tn) acc[tm][tn] = v8f_zero();

  // per-thread staging coordinates: 2 x 16B per matrix per tile
  const int r0s = t >> 2, c0s = (t & 3) * 8;             // vi = t
  const int r1s = (t + 256) >> 2, c1s = ((t + 256) & 3) * 8;

  auto stage = [&](int kb, int buf) {
    async_copy16(&sA[buf][r0s][c0s], &A[(size_t)(m0 + r0s) * K + kb * 32 + c0s]);
    async_copy16(&sB[buf][r0s][c0s], &Bt[(size_t)(n0 + r0s) * K + kb * 32 + c0s]);
    async_copy16(&sA[buf][r1s][c1s], &A[(size_t)(m0 + r1s) * K + kb * 32 + c1s]);
    async_copy16(&sB[buf][r1s][c1s], &Bt[(size_t)(n0 + r1s) * K + kb * 32 + c1s]);
  };

  const int nk = K >> 5;
  stage(0, 0);
  wait_async0();
  __syncthreads();

  for (int kb = 0; kb < nk; ++kb) {
    const int cur = kb & 1;
    if (kb + 1 < nk) stage(kb + 1, cur ^ 1);   // overlap with WMMA below
    if (kb + 2 < nk) {                         // GL2 prefetch two tiles ahead
      __builtin_prefetch(&A[(size_t)(m0 + (t >> 1)) * K + (kb + 2) * 32 + (t & 1) * 16], 0, 2);
      __builtin_prefetch(&Bt[(size_t)(n0 + (t >> 1)) * K + (kb + 2) * 32 + (t & 1) * 16], 0, 2);
    }

    v16bf aF[4], bF[2];
#pragma unroll
    for (int tm = 0; tm < 4; ++tm)
      aF[tm] = *(const v16bf*)&sA[cur][wm * 64 + tm * 16 + lr][lh * 16];
#pragma unroll
    for (int tn = 0; tn < 2; ++tn)
      bF[tn] = *(const v16bf*)&sB[cur][wn * 32 + tn * 16 + lr][lh * 16];
#pragma unroll
    for (int tm = 0; tm < 4; ++tm)
#pragma unroll
      for (int tn = 0; tn < 2; ++tn)
        acc[tm][tn] = wmma_bf16(aF[tm], bF[tn], acc[tm][tn]);

    wait_async0();      // next tile fully landed in LDS
    __syncthreads();    // visible to all waves; prior buffer reads done
  }

  // epilogue: bias + residual fused
#pragma unroll
  for (int tm = 0; tm < 4; ++tm)
#pragma unroll
    for (int tn = 0; tn < 2; ++tn) {
      const int col = n0 + wn * 32 + tn * 16 + lr;
      const float bv = bias ? bias[col] : 0.0f;
#pragma unroll
      for (int i = 0; i < 8; ++i) {
        const int row = m0 + wm * 64 + tm * 16 + lh * 8 + i;
        float v = acc[tm][tn][i] + bv;
        if (res) v += res[(size_t)row * N + col];
        C[(size_t)row * N + col] = v;
      }
    }
}

// ---------------------------------------------------------------------------
// Flash attention (non-causal), WMMA for QK^T and PV, online softmax.
// grid.x = B*H, grid.y = S/128. 8 waves; wave w owns 16 query rows.
// Softmax scale 1/sqrt(64) folded into Q at staging time.
// ---------------------------------------------------------------------------
__global__ __launch_bounds__(256) void k_attention(const float* __restrict__ qkv,
                                                   bf16_t* __restrict__ ctx) {
  __shared__ bf16_t sQ[128][64];       // 16 KB
  __shared__ bf16_t sK[64][64];        //  8 KB  [t][d]
  __shared__ bf16_t sVt[64][64];       //  8 KB  [v][t]
  __shared__ bf16_t sP[8][16][64];     // 16 KB  per-wave P tiles

  const int t = threadIdx.x, lane = t & 31, w = t >> 5;
  const int b = blockIdx.x >> 4, h = blockIdx.x & 15;
  const int row0 = blockIdx.y * 128;
  const int qr0 = w * 16;
  const int lr = lane & 15, lh = lane >> 4;

  // stage Q, scaled by 1/8
#pragma unroll
  for (int j = 0; j < 8; ++j) {
    int f4 = t + j * 256;              // 2048 float4 loads
    int r = f4 >> 4, c = (f4 & 15) << 2;
    const float4 q4 =
        *(const float4*)&qkv[((size_t)(b * S_LEN + row0 + r)) * QKV_OUT + h * 64 + c];
    sQ[r][c + 0] = to_bf16(q4.x * 0.125f);
    sQ[r][c + 1] = to_bf16(q4.y * 0.125f);
    sQ[r][c + 2] = to_bf16(q4.z * 0.125f);
    sQ[r][c + 3] = to_bf16(q4.w * 0.125f);
  }

  v8f accO[4];
  float m[8], l[8];
#pragma unroll
  for (int tn = 0; tn < 4; ++tn) accO[tn] = v8f_zero();
#pragma unroll
  for (int i = 0; i < 8; ++i) { m[i] = -1e30f; l[i] = 0.0f; }

  for (int kb = 0; kb < S_LEN / 64; ++kb) {
    const int t0 = kb * 64;
#pragma unroll
    for (int j = 0; j < 4; ++j) {
      int f4 = t + j * 256;            // 1024 float4 loads per matrix
      int r = f4 >> 4, c = (f4 & 15) << 2;
      const size_t base = ((size_t)(b * S_LEN + t0 + r)) * QKV_OUT + h * 64;
      const float4 k4 = *(const float4*)&qkv[base + 1024 + c];
      sK[r][c + 0] = to_bf16(k4.x); sK[r][c + 1] = to_bf16(k4.y);
      sK[r][c + 2] = to_bf16(k4.z); sK[r][c + 3] = to_bf16(k4.w);
      const float4 v4 = *(const float4*)&qkv[base + 2048 + c];
      sVt[c + 0][r] = to_bf16(v4.x); sVt[c + 1][r] = to_bf16(v4.y);
      sVt[c + 2][r] = to_bf16(v4.z); sVt[c + 3][r] = to_bf16(v4.w);
    }
    __syncthreads();

    // S = (Q/8) * K^T : 16x64 per wave, K-dim 64 -> 2 wmma steps x 4 tiles
    v8f s[4];
#pragma unroll
    for (int tn = 0; tn < 4; ++tn) s[tn] = v8f_zero();
#pragma unroll
    for (int kk = 0; kk < 2; ++kk) {
      const v16bf aq = *(const v16bf*)&sQ[qr0 + lr][kk * 32 + lh * 16];
#pragma unroll
      for (int tn = 0; tn < 4; ++tn) {
        const v16bf bk = *(const v16bf*)&sK[tn * 16 + lr][kk * 32 + lh * 16];
        s[tn] = wmma_bf16(aq, bk, s[tn]);
      }
    }

    // online softmax in C-layout registers; row = lh*8 + i
#pragma unroll
    for (int i = 0; i < 8; ++i) {
      float v = fmaxf(fmaxf(s[0][i], s[1][i]), fmaxf(s[2][i], s[3][i]));
      v = fmaxf(v, __shfl_xor(v, 8, 32));
      v = fmaxf(v, __shfl_xor(v, 4, 32));
      v = fmaxf(v, __shfl_xor(v, 2, 32));
      v = fmaxf(v, __shfl_xor(v, 1, 32));
      const float mn = fmaxf(m[i], v);
      const float corr = __expf(m[i] - mn);
      m[i] = mn;
      float sum = 0.0f;
#pragma unroll
      for (int tn = 0; tn < 4; ++tn) {
        float p = __expf(s[tn][i] - mn);
        s[tn][i] = p;
        sum += p;
      }
      sum += __shfl_xor(sum, 8, 32);
      sum += __shfl_xor(sum, 4, 32);
      sum += __shfl_xor(sum, 2, 32);
      sum += __shfl_xor(sum, 1, 32);
      l[i] = l[i] * corr + sum;
#pragma unroll
      for (int tn = 0; tn < 4; ++tn) accO[tn][i] *= corr;
      const int pr = (lh << 3) + i;
#pragma unroll
      for (int tn = 0; tn < 4; ++tn)
        sP[w][pr][tn * 16 + lr] = to_bf16(s[tn][i]);
    }
    __syncthreads();

    // O += P * V : reduction over t (64) -> 2 wmma steps x 4 output tiles
#pragma unroll
    for (int kk = 0; kk < 2; ++kk) {
      const v16bf ap = *(const v16bf*)&sP[w][lr][kk * 32 + lh * 16];
#pragma unroll
      for (int tn = 0; tn < 4; ++tn) {
        const v16bf bv = *(const v16bf*)&sVt[tn * 16 + lr][kk * 32 + lh * 16];
        accO[tn] = wmma_bf16(ap, bv, accO[tn]);
      }
    }
    __syncthreads();
  }

  // normalize and write ctx as bf16 [BS][H*64]
#pragma unroll
  for (int i = 0; i < 8; ++i) {
    const float inv = 1.0f / l[i];
    const int row = row0 + qr0 + (lh << 3) + i;
    const size_t base = ((size_t)(b * S_LEN + row)) * D_EMB + h * 64;
#pragma unroll
    for (int tn = 0; tn < 4; ++tn)
      ctx[base + tn * 16 + lr] = to_bf16(accO[tn][i] * inv);
  }
}

// ---------------------------------------------------------------------------
// Host orchestration
// ---------------------------------------------------------------------------
extern "C" void kernel_launch(void* const* d_in, const int* in_sizes, int n_in,
                              void* d_out, int out_size, void* d_ws, size_t ws_size,
                              hipStream_t stream) {
  (void)in_sizes; (void)n_in; (void)out_size; (void)ws_size;
  const int*   tokens   = (const int*)d_in[0];
  const float* emb      = (const float*)d_in[1];
  const float* qkv_w    = (const float*)d_in[2];
  const float* qkv_b    = (const float*)d_in[3];
  const float* unproj_w = (const float*)d_in[4];
  const float* unproj_b = (const float*)d_in[5];
  const float* bank_w   = (const float*)d_in[6];
  const float* bank_b   = (const float*)d_in[7];

  char* ws = (char*)d_ws;
  size_t off = 0;
  auto alloc = [&](size_t bytes) -> void* {
    void* p = ws + off;
    off = (off + bytes + 255) & ~(size_t)255;
    return p;
  };

  bf16_t* embbf    = (bf16_t*)alloc((size_t)N_VOCAB * D_EMB * 2);
  bf16_t* qkvwt    = (bf16_t*)alloc((size_t)4 * QKV_OUT * D_EMB * 2);
  bf16_t* unprojwt = (bf16_t*)alloc((size_t)D_EMB * D_EMB * 2);
  bf16_t* bankwt   = (bf16_t*)alloc((size_t)4 * D_EMB * D_EMB * 2);
  float*  x0       = (float*)alloc((size_t)BS_TOK * D_EMB * 4);
  float*  x1       = (float*)alloc((size_t)BS_TOK * D_EMB * 4);
  bf16_t* xbf      = (bf16_t*)alloc((size_t)BS_TOK * D_EMB * 2);
  float*  qkvbuf   = (float*)alloc((size_t)BS_TOK * QKV_OUT * 4);
  bf16_t* qbf      = (bf16_t*)alloc((size_t)BS_TOK * D_EMB * 2);
  bf16_t* ctxbf    = (bf16_t*)alloc((size_t)BS_TOK * D_EMB * 2);
  float*  tmp      = (float*)alloc((size_t)BS_TOK * D_EMB * 4);

  // --- one-time precision/layout conversions ---
  k_convert_bf16<<<4096, 256, 0, stream>>>(emb, embbf, (size_t)N_VOCAB * D_EMB);
  for (int i = 0; i < 4; ++i) {
    k_transpose_bf16<<<dim3(QKV_OUT / 32, D_EMB / 32), 256, 0, stream>>>(
        qkv_w + (size_t)i * D_EMB * QKV_OUT, qkvwt + (size_t)i * QKV_OUT * D_EMB,
        D_EMB, QKV_OUT);
    k_transpose_bf16<<<dim3(D_EMB / 32, D_EMB / 32), 256, 0, stream>>>(
        bank_w + (size_t)i * D_EMB * D_EMB, bankwt + (size_t)i * D_EMB * D_EMB,
        D_EMB, D_EMB);
  }
  k_transpose_bf16<<<dim3(D_EMB / 32, D_EMB / 32), 256, 0, stream>>>(
      unproj_w, unprojwt, D_EMB, D_EMB);

  // --- embedding ---
  k_embed<<<BS_TOK, 256, 0, stream>>>(tokens, emb, x0, xbf);

  float* xin = x0;
  float* xout = x1;
  for (int i = 0; i < 4; ++i) {
    // qkv = xbf @ qkv_w[i] + qkv_b[i]
    k_gemm_bf16<<<dim3(QKV_OUT / 128, BS_TOK / 128), 256, 0, stream>>>(
        xbf, qkvwt + (size_t)i * QKV_OUT * D_EMB, qkv_b + (size_t)i * QKV_OUT,
        nullptr, qkvbuf, BS_TOK, QKV_OUT, D_EMB);
    // q slice -> bf16 for bank GEMM
    k_extract_q<<<4096, 256, 0, stream>>>(qkvbuf, qbf);
    // attention -> ctx (bf16)
    k_attention<<<dim3(2 * H_NUM, S_LEN / 128), 256, 0, stream>>>(qkvbuf, ctxbf);
    // tmp = x + q @ bank_w[i] + bank_b[i]
    k_gemm_bf16<<<dim3(D_EMB / 128, BS_TOK / 128), 256, 0, stream>>>(
        qbf, bankwt + (size_t)i * D_EMB * D_EMB, bank_b + (size_t)i * D_EMB,
        xin, tmp, BS_TOK, D_EMB, D_EMB);
    // x' = tmp + ctx @ unproj_w + unproj_b
    k_gemm_bf16<<<dim3(D_EMB / 128, BS_TOK / 128), 256, 0, stream>>>(
        ctxbf, unprojwt, unproj_b, tmp, xout, BS_TOK, D_EMB, D_EMB);
    // x' -> bf16 for next layer / logits
    k_convert_bf16<<<4096, 256, 0, stream>>>(xout, xbf, (size_t)BS_TOK * D_EMB);
    float* sw = xin; xin = xout; xout = sw;
  }

  // logits = xbf @ emb^T  (embbf is already [vocab][d] = Bt layout)
  k_gemm_bf16<<<dim3(N_VOCAB / 128, BS_TOK / 128), 256, 0, stream>>>(
      xbf, embbf, nullptr, nullptr, (float*)d_out, BS_TOK, N_VOCAB, D_EMB);
}
